// PCN_encoder_73177652789698
// MI455X (gfx1250) — matrix operand order repaired and checked by reading, not compiled, not run on hardware
//
#include <hip/hip_runtime.h>
#include <hip/hip_bf16.h>

typedef _Float16 f16;
typedef __attribute__((ext_vector_type(16))) _Float16 v16h;
typedef __attribute__((ext_vector_type(8)))  _Float16 v8h;
typedef __attribute__((ext_vector_type(2)))  _Float16 v2h;
typedef __attribute__((ext_vector_type(8)))  float    v8f;

#define BB   16
#define NP   2048
#define CSA  512
#define COUT 1024
#define NEG_INF (-3.0e38f)

// ---------------------------------------------------------------------------
// WMMA fragment helpers (layouts per CDNA5 ISA 7.12.2, wave32)
// A: 16x32 f16, element (m,k). lanes 0-15: m=lane, K base 0; lanes 16-31: +8 / +16+8
// B: 32x16 f16, element (k,n). lanes 0-15: n=lane, K base 0; lanes 16-31: +8 / +16+8
// C/D: 16x16 f32. lane: n = l&15 ; VGPR v: m = v + 8*(l>>4)
// ---------------------------------------------------------------------------
__device__ __forceinline__ v16h frag_a(const f16* A, int lda, int m0, int k0) {
    int l = threadIdx.x & 31;
    int m = m0 + (l & 15);
    int kb = k0 + ((l >> 4) << 3);          // +8 for upper half-wave
    const f16* p = A + (size_t)m * lda + kb;
    v16h a;
#pragma unroll
    for (int v = 0; v < 4; ++v) {
        v2h x = *(const v2h*)(p + 2 * v);          // K = kb + 2v, +1
        a[2 * v]     = x[0];
        a[2 * v + 1] = x[1];
        v2h y = *(const v2h*)(p + 16 + 2 * v);     // K = kb + 16 + 2v, +1
        a[8 + 2 * v]     = y[0];
        a[8 + 2 * v + 1] = y[1];
    }
    return a;
}

__device__ __forceinline__ v16h frag_b(const f16* Bm, int ldb, int n0, int k0) {
    int l = threadIdx.x & 31;
    int n = n0 + (l & 15);
    int kb = k0 + ((l >> 4) << 3);          // +8 for upper half-wave
    const f16* p = Bm + (size_t)n * ldb + kb;
    v16h b;
#pragma unroll
    for (int v = 0; v < 4; ++v) {
        v2h x = *(const v2h*)(p + 2 * v);
        b[2 * v]     = x[0];
        b[2 * v + 1] = x[1];
        v2h y = *(const v2h*)(p + 16 + 2 * v);
        b[8 + 2 * v]     = y[0];
        b[8 + 2 * v + 1] = y[1];
    }
    return b;
}

// Multi-accumulator: one A fragment feeds NT back-to-back WMMAs (16 x 16*NT strip)
template <int NT>
__device__ __forceinline__ void gemm_k_multi(const f16* A, int lda, int m0,
                                             const f16* Bm, int ldb, int n0,
                                             int K, v8f* acc) {
    for (int k0 = 0; k0 < K; k0 += 32) {
        v16h a = frag_a(A, lda, m0, k0);
#pragma unroll
        for (int t = 0; t < NT; ++t) {
            v16h b = frag_b(Bm, ldb, n0 + 16 * t, k0);
            acc[t] = __builtin_amdgcn_wmma_f32_16x16x32_f16(false, a, false, b,
                                                            (short)0, acc[t],
                                                            false, false);
        }
    }
}

// Channel-last f16 packed store: out[n*ldo + m], 8 consecutive m -> one b128 store
__device__ __forceinline__ void store_cl(f16* out, int ldo, int m0, int n0,
                                         v8f acc, const float* bias, int relu) {
    int l = threadIdx.x & 31;
    int n = n0 + (l & 15);
    int mb = m0 + ((l >> 4) << 3);
    v8h pk;
#pragma unroll
    for (int v = 0; v < 8; ++v) {
        float x = acc[v];
        if (bias) x += bias[mb + v];
        if (relu) x = x > 0.f ? x : 0.f;
        pk[v] = (f16)x;
    }
    *(v8h*)(out + (size_t)n * ldo + mb) = pk;
}

// ---------------------------------------------------------------------------
// Weight fp32 -> f16 (optionally column-sliced)
// ---------------------------------------------------------------------------
__global__ void k_cvt(const float* __restrict__ src, f16* __restrict__ dst,
                      int rows, int cols, int src_ld) {
    int i = blockIdx.x * blockDim.x + threadIdx.x;
    if (i >= rows * cols) return;
    int r = i / cols, c = i - r * cols;
    dst[i] = (f16)src[(size_t)r * src_ld + c];
}

// ---------------------------------------------------------------------------
// Positional encoding + conv1 (21 -> 128) + relu. h1 channel-last f16 [B][N][128]
// ---------------------------------------------------------------------------
__global__ __launch_bounds__(256)
void k_posenc_conv1(const float* __restrict__ x, const float* __restrict__ w1,
                    const float* __restrict__ b1, f16* __restrict__ h1) {
    __shared__ float sw[128 * 21 + 128];
    for (int i = threadIdx.x; i < 128 * 21; i += blockDim.x) sw[i] = w1[i];
    for (int i = threadIdx.x; i < 128; i += blockDim.x) sw[128 * 21 + i] = b1[i];
    __syncthreads();
    int idx = blockIdx.x * blockDim.x + threadIdx.x;   // b*N + n
    if (idx >= BB * NP) return;
    int b = idx / NP, n = idx - b * NP;
    float e[21];
#pragma unroll
    for (int c = 0; c < 3; ++c) {
        float t = x[((size_t)b * 3 + c) * NP + n];
        e[c]      = t;
        e[3 + c]  = __sinf(t);
        e[6 + c]  = __cosf(t);
        e[9 + c]  = __sinf(2.f * t);
        e[12 + c] = __cosf(2.f * t);
        e[15 + c] = __sinf(4.f * t);
        e[18 + c] = __cosf(4.f * t);
    }
    f16* o = h1 + (size_t)idx * 128;
    for (int oc = 0; oc < 128; ++oc) {
        float acc = sw[128 * 21 + oc];
        const float* wr = &sw[oc * 21];
#pragma unroll
        for (int c = 0; c < 21; ++c) acc += wr[c] * e[c];
        o[oc] = (f16)(acc > 0.f ? acc : 0.f);
    }
}

// ---------------------------------------------------------------------------
// Generic channel-last WMMA GEMM: Out[b][n][m] = act( A . B + bias )
// grid = B * Mtiles * 8 ; block = 128 (4 waves); each wave owns a 16x64 strip
// ---------------------------------------------------------------------------
__global__ __launch_bounds__(128)
void k_gemm_cl(const f16* __restrict__ A, long long Abs, int lda,
               const f16* __restrict__ Bb, long long Bbs, int ldb,
               int K, int Mtiles,
               const float* __restrict__ bias, long long biasbs, int relu,
               f16* __restrict__ Ob, long long Obs, int ldo) {
    int wave = threadIdx.x >> 5;
    int ng = blockIdx.x % 8;
    int mt = (blockIdx.x / 8) % Mtiles;
    int b  = (blockIdx.x / 8) / Mtiles;
    int m0 = mt * 16;
    int n0 = (ng * 4 + wave) * 64;
    const f16* Ab = A + (size_t)b * Abs;
    const f16* Bm = Bb + (size_t)b * Bbs;
    v8f acc[4] = {};
    gemm_k_multi<4>(Ab, lda, m0, Bm, ldb, n0, K, acc);
    const float* bp = bias ? bias + (size_t)b * biasbs : nullptr;
    f16* Om = Ob + (size_t)b * Obs;
#pragma unroll
    for (int t = 0; t < 4; ++t) store_cl(Om, ldo, m0, n0 + 16 * t, acc[t], bp, relu);
}

// ---------------------------------------------------------------------------
// xv conv: channel-major f16 output [B][512][N]
// ---------------------------------------------------------------------------
__global__ __launch_bounds__(128)
void k_gemm_xv(const f16* __restrict__ W, const float* __restrict__ bias,
               const f16* __restrict__ Bb, f16* __restrict__ Ob) {
    int wave = threadIdx.x >> 5;
    int ng = blockIdx.x % 8;
    int mt = (blockIdx.x / 8) % 32;
    int b  = (blockIdx.x / 8) / 32;
    int m0 = mt * 16;
    int n0 = (ng * 4 + wave) * 64;
    const f16* Bm = Bb + (size_t)b * NP * CSA;
    v8f acc[4] = {};
    gemm_k_multi<4>(W, CSA, m0, Bm, CSA, n0, CSA, acc);
    int l = threadIdx.x & 31;
    int mb = m0 + ((l >> 4) << 3);
    f16* o = Ob + (size_t)b * CSA * NP;
#pragma unroll
    for (int t = 0; t < 4; ++t) {
        int n = n0 + 16 * t + (l & 15);
#pragma unroll
        for (int v = 0; v < 8; ++v)
            o[(size_t)(mb + v) * NP + n] = (f16)(acc[t][v] + bias[mb + v]);
    }
}

// ---------------------------------------------------------------------------
// g = max over n of h2 (pre-activation)   g[b][o]
// ---------------------------------------------------------------------------
__global__ void k_gmax(const f16* __restrict__ h2, float* __restrict__ g) {
    int i = blockIdx.x * blockDim.x + threadIdx.x;
    if (i >= BB * 256) return;
    int b = i / 256, o = i - b * 256;
    const f16* p = h2 + (size_t)b * NP * 256 + o;
    float m = NEG_INF;
    for (int n = 0; n < NP; ++n) {
        float v = (float)p[(size_t)n * 256];
        m = v > m ? v : m;
    }
    g[i] = m;
}

// gvec[b][oo] = b3[oo] + w3[oo][256:512] . g[b]    (folds global-feature half of conv3)
__global__ void k_gvec(const float* __restrict__ w3, const float* __restrict__ b3,
                       const float* __restrict__ g, float* __restrict__ gvec) {
    int i = blockIdx.x * blockDim.x + threadIdx.x;
    if (i >= BB * CSA) return;
    int b = i / CSA, oo = i - b * CSA;
    const float* wr = w3 + (size_t)oo * CSA + 256;
    const float* gb = g + b * 256;
    float acc = b3[oo];
    for (int c = 0; c < 256; ++c) acc += wr[c] * gb[c];
    gvec[i] = acc;
}

// ---------------------------------------------------------------------------
// Row softmax over keys, f16 in place (attn[b][q][k])
// ---------------------------------------------------------------------------
__global__ __launch_bounds__(256)
void k_softmax(f16* __restrict__ attn) {
    __shared__ float red[256];
    int b = blockIdx.x / NP, q = blockIdx.x - b * NP;
    f16* row = attn + ((size_t)b * NP + q) * NP;
    int t = threadIdx.x;
    float mx = NEG_INF;
    for (int k = t; k < NP; k += 256) mx = fmaxf(mx, (float)row[k]);
    red[t] = mx; __syncthreads();
    for (int s = 128; s > 0; s >>= 1) {
        if (t < s) red[t] = fmaxf(red[t], red[t + s]);
        __syncthreads();
    }
    mx = red[0]; __syncthreads();
    float sum = 0.f;
    for (int k = t; k < NP; k += 256) sum += __expf((float)row[k] - mx);
    red[t] = sum; __syncthreads();
    for (int s = 128; s > 0; s >>= 1) {
        if (t < s) red[t] += red[t + s];
        __syncthreads();
    }
    float inv = 1.f / red[0];
    for (int k = t; k < NP; k += 256) row[k] = (f16)(__expf((float)row[k] - mx) * inv);
}

// Column L1 renorm: each thread owns key column k of batch b
__global__ void k_renorm(f16* __restrict__ attn) {
    int i = blockIdx.x * blockDim.x + threadIdx.x;
    if (i >= BB * NP) return;
    int b = i / NP, k = i - b * NP;
    f16* base = attn + (size_t)b * NP * NP + k;
    float s = 0.f;
    for (int q = 0; q < NP; ++q) s += (float)base[(size_t)q * NP];
    float inv = 1.f / (1e-9f + s);
    for (int q = 0; q < NP; ++q) {
        size_t off = (size_t)q * NP;
        base[off] = (f16)((float)base[off] * inv);
    }
}

// ---------------------------------------------------------------------------
// x_r GEMM (M=c, N=q, K=keys) fused with hmr = h3 - x_r, channel-last f16 out
// A = xv channel-major [c][k], B = attn rows [q][k]
// ---------------------------------------------------------------------------
__global__ __launch_bounds__(128)
void k_xr(const f16* __restrict__ attn, const f16* __restrict__ xvt,
          const f16* __restrict__ h3, f16* __restrict__ hmr) {
    int wave = threadIdx.x >> 5;
    int ng = blockIdx.x % 8;
    int mt = (blockIdx.x / 8) % 32;
    int b  = (blockIdx.x / 8) / 32;
    int m0 = mt * 16;                    // channel tile
    int n0 = (ng * 4 + wave) * 64;       // query strip
    const f16* A  = xvt + (size_t)b * CSA * NP;
    const f16* Bm = attn + (size_t)b * NP * NP;
    v8f acc[4] = {};
    gemm_k_multi<4>(A, NP, m0, Bm, NP, n0, NP, acc);
    int l = threadIdx.x & 31;
    int cb = m0 + ((l >> 4) << 3);
#pragma unroll
    for (int t = 0; t < 4; ++t) {
        int q = n0 + 16 * t + (l & 15);
        size_t off = ((size_t)b * NP + q) * CSA + cb;
        v8h hv = *(const v8h*)(h3 + off);
        v8h pk;
#pragma unroll
        for (int v = 0; v < 8; ++v) pk[v] = (f16)((float)hv[v] - acc[t][v]);
        *(v8h*)(hmr + off) = pk;
    }
}

// ---------------------------------------------------------------------------
// d = wt . hmr + bt, fp32 channel-major [B][512][N] (coalesced BN stats later)
// ---------------------------------------------------------------------------
__global__ __launch_bounds__(128)
void k_convt(const f16* __restrict__ W, const float* __restrict__ bt,
             const f16* __restrict__ hmr, float* __restrict__ d) {
    int wave = threadIdx.x >> 5;
    int ng = blockIdx.x % 8;
    int mt = (blockIdx.x / 8) % 32;
    int b  = (blockIdx.x / 8) / 32;
    int m0 = mt * 16;
    int n0 = (ng * 4 + wave) * 64;
    const f16* Bm = hmr + (size_t)b * NP * CSA;
    v8f acc[4] = {};
    gemm_k_multi<4>(W, CSA, m0, Bm, CSA, n0, CSA, acc);
    int l = threadIdx.x & 31;
    int mb = m0 + ((l >> 4) << 3);
    float* o = d + (size_t)b * CSA * NP;
#pragma unroll
    for (int t = 0; t < 4; ++t) {
        int n = n0 + 16 * t + (l & 15);
#pragma unroll
        for (int v = 0; v < 8; ++v)
            o[(size_t)(mb + v) * NP + n] = acc[t][v] + bt[mb + v];
    }
}

// BatchNorm batch statistics per channel (over B and N)
__global__ __launch_bounds__(256)
void k_bnstats(const float* __restrict__ d, float* __restrict__ mu,
               float* __restrict__ rsig) {
    __shared__ float s1[256], s2[256];
    int o = blockIdx.x;
    int t = threadIdx.x;
    float a = 0.f, sq = 0.f;
    for (int b = 0; b < BB; ++b) {
        const float* p = d + ((size_t)b * CSA + o) * NP;
        for (int n = t; n < NP; n += 256) {
            float v = p[n];
            a += v; sq += v * v;
        }
    }
    s1[t] = a; s2[t] = sq; __syncthreads();
    for (int s = 128; s > 0; s >>= 1) {
        if (t < s) { s1[t] += s1[t + s]; s2[t] += s2[t + s]; }
        __syncthreads();
    }
    if (t == 0) {
        float inv_n = 1.f / (float)(BB * NP);
        float m = s1[0] * inv_n;
        float var = s2[0] * inv_n - m * m;
        mu[o] = m;
        rsig[o] = rsqrtf(var + 1e-5f);
    }
}

// hfin = h3 + relu(gamma*(d-mu)*rsig + beta), channel-last f16.
// 64x64 LDS transpose tile -> both the fp32 reads and f16 writes are coalesced.
__global__ __launch_bounds__(256)
void k_hfin(const float* __restrict__ d, const f16* __restrict__ h3,
            const float* __restrict__ mu, const float* __restrict__ rsig,
            const float* __restrict__ gamma, const float* __restrict__ beta,
            f16* __restrict__ hfin) {
    __shared__ float tile[64][65];
    int bo = blockIdx.x % 8;            // channel tile (512/64)
    int bn = (blockIdx.x / 8) % 32;     // point tile (2048/64)
    int b  = blockIdx.x / (8 * 32);
    int o0 = bo * 64, n0 = bn * 64;
    for (int idx = threadIdx.x; idx < 64 * 64; idx += 256) {
        int i = idx >> 6, j = idx & 63;
        tile[i][j] = d[((size_t)b * CSA + o0 + i) * NP + n0 + j];
    }
    __syncthreads();
    for (int idx = threadIdx.x; idx < 64 * 64; idx += 256) {
        int j = idx >> 6;               // point-local
        int i = idx & 63;               // channel-local
        int o = o0 + i, n = n0 + j;
        float t = gamma[o] * (tile[i][j] - mu[o]) * rsig[o] + beta[o];
        t = t > 0.f ? t : 0.f;
        size_t off = ((size_t)b * NP + n) * CSA + o;
        hfin[off] = (f16)((float)h3[off] + t);
    }
}

// ---------------------------------------------------------------------------
// conv4 (512 -> 1024) fused with max over N. One block per (b, 16-j tile).
// Weight tile is DMA'd into LDS with GLOBAL_LOAD_ASYNC_TO_LDS_B128 (ASYNCcnt),
// then re-read 8x per wave as ds_load fragments.
// ---------------------------------------------------------------------------
__global__ __launch_bounds__(128)
void k_conv4max(const f16* __restrict__ W, const float* __restrict__ b4,
                const f16* __restrict__ hfin, float* __restrict__ out) {
    __shared__ __align__(16) f16 wtile[16 * CSA];    // 16 KB weight tile
    __shared__ float lmax[4][16][16];
    int wave = threadIdx.x >> 5, lane = threadIdx.x & 31;
    int jt = blockIdx.x % 64, b = blockIdx.x / 64;
    int m0 = jt * 16;

    // Async-stage W[m0:m0+16][0:512] into LDS: 16384 B = 1024 x b128 chunks
    {
        const f16* gsrc = W + (size_t)m0 * CSA;
        unsigned ldsbase = (unsigned)(uintptr_t)&wtile[0];   // LDS aperture: low 32 bits
        for (int it = threadIdx.x; it < 1024; it += 128) {
            const f16* gp = gsrc + it * 8;
            unsigned lo = ldsbase + it * 16;
            asm volatile("global_load_async_to_lds_b128 %0, %1, off"
                         :: "v"(lo), "v"(gp) : "memory");
        }
        asm volatile("s_wait_asynccnt 0x0" ::: "memory");
    }
    __syncthreads();

    const f16* Bm = hfin + (size_t)b * NP * CSA;
    v8f rmax;
#pragma unroll
    for (int v = 0; v < 8; ++v) rmax[v] = NEG_INF;
    for (int g = wave; g < 32; g += 4) {             // 32 strips of 64 points
        v8f acc[4] = {};
        gemm_k_multi<4>(wtile, CSA, 0, Bm, CSA, g * 64, CSA, acc);
#pragma unroll
        for (int t = 0; t < 4; ++t)
#pragma unroll
            for (int v = 0; v < 8; ++v) rmax[v] = fmaxf(rmax[v], acc[t][v]);
    }
    int n16 = lane & 15, half = lane >> 4;
#pragma unroll
    for (int v = 0; v < 8; ++v) lmax[wave][half * 8 + v][n16] = rmax[v];
    __syncthreads();
    if (threadIdx.x < 16) {
        int m = threadIdx.x;
        float mx = NEG_INF;
        for (int w = 0; w < 4; ++w)
            for (int n = 0; n < 16; ++n) mx = fmaxf(mx, lmax[w][m][n]);
        out[(size_t)b * COUT + m0 + m] = mx + b4[m0 + m];
    }
}

// ---------------------------------------------------------------------------
extern "C" void kernel_launch(void* const* d_in, const int* in_sizes, int n_in,
                              void* d_out, int out_size, void* d_ws, size_t ws_size,
                              hipStream_t stream) {
    const float* x    = (const float*)d_in[0];
    const float* w1   = (const float*)d_in[1];
    const float* b1   = (const float*)d_in[2];
    const float* w2   = (const float*)d_in[3];
    const float* b2   = (const float*)d_in[4];
    const float* w3   = (const float*)d_in[5];
    const float* b3   = (const float*)d_in[6];
    const float* w4   = (const float*)d_in[7];
    const float* b4   = (const float*)d_in[8];
    const float* wqk  = (const float*)d_in[9];
    const float* wv   = (const float*)d_in[10];
    const float* bv   = (const float*)d_in[11];
    const float* wt   = (const float*)d_in[12];
    const float* bt   = (const float*)d_in[13];
    const float* gam  = (const float*)d_in[14];
    const float* bet  = (const float*)d_in[15];
    float* out = (float*)d_out;

    // ---- workspace layout ----
    char* w = (char*)d_ws;
    size_t off = 0;
    auto alloc = [&](size_t bytes) {
        size_t r = off;
        off = (off + bytes + 255) & ~(size_t)255;
        return r;
    };
    f16*   w2h  = (f16*)(w + alloc((size_t)256 * 128 * 2));
    f16*   w3ah = (f16*)(w + alloc((size_t)512 * 256 * 2));
    f16*   wqkh = (f16*)(w + alloc((size_t)128 * 512 * 2));
    f16*   wvh  = (f16*)(w + alloc((size_t)512 * 512 * 2));
    f16*   wth  = (f16*)(w + alloc((size_t)512 * 512 * 2));
    f16*   w4h  = (f16*)(w + alloc((size_t)1024 * 512 * 2));
    f16*   h1   = (f16*)(w + alloc((size_t)BB * NP * 128 * 2));
    f16*   h2   = (f16*)(w + alloc((size_t)BB * NP * 256 * 2));
    float* g    = (float*)(w + alloc((size_t)BB * 256 * 4));
    float* gvec = (float*)(w + alloc((size_t)BB * CSA * 4));
    f16*   h3   = (f16*)(w + alloc((size_t)BB * NP * CSA * 2));
    f16*   xqk  = (f16*)(w + alloc((size_t)BB * NP * 128 * 2));
    f16*   xvt  = (f16*)(w + alloc((size_t)BB * CSA * NP * 2));
    f16*   attn = (f16*)(w + alloc((size_t)BB * NP * NP * 2));   // 134 MB
    f16*   hmr  = (f16*)(w + alloc((size_t)BB * NP * CSA * 2));
    float* mu   = (float*)(w + alloc((size_t)CSA * 4));
    float* rsig = (float*)(w + alloc((size_t)CSA * 4));
    float* dbuf = (float*)attn;   // reuse: attn dead after k_xr, 67 MB <= 134 MB
    f16*   hfin = hmr;            // reuse: hmr dead after k_convt

    // ---- weight conversions fp32 -> f16 ----
    k_cvt<<<(256 * 128 + 255) / 256, 256, 0, stream>>>(w2, w2h, 256, 128, 128);
    k_cvt<<<(512 * 256 + 255) / 256, 256, 0, stream>>>(w3, w3ah, 512, 256, 512);
    k_cvt<<<(128 * 512 + 255) / 256, 256, 0, stream>>>(wqk, wqkh, 128, 512, 512);
    k_cvt<<<(512 * 512 + 255) / 256, 256, 0, stream>>>(wv, wvh, 512, 512, 512);
    k_cvt<<<(512 * 512 + 255) / 256, 256, 0, stream>>>(wt, wth, 512, 512, 512);
    k_cvt<<<(1024 * 512 + 255) / 256, 256, 0, stream>>>(w4, w4h, 1024, 512, 512);

    // ---- stage 1: posenc + conv1 + relu ----
    k_posenc_conv1<<<(BB * NP + 255) / 256, 256, 0, stream>>>(x, w1, b1, h1);

    // ---- stage 2: conv2 (128 -> 256), no activation ----
    k_gemm_cl<<<BB * 16 * 8, 128, 0, stream>>>(
        w2h, 0, 128, h1, (long long)NP * 128, 128, 128, 16,
        b2, 0, 0, h2, (long long)NP * 256, 256);

    // ---- global feature + folded conv3 bias ----
    k_gmax<<<(BB * 256 + 255) / 256, 256, 0, stream>>>(h2, g);
    k_gvec<<<(BB * CSA + 255) / 256, 256, 0, stream>>>(w3, b3, g, gvec);

    // ---- stage 3: conv3 local half (256 -> 512) + gvec bias + relu ----
    k_gemm_cl<<<BB * 32 * 8, 128, 0, stream>>>(
        w3ah, 0, 256, h2, (long long)NP * 256, 256, 256, 32,
        gvec, CSA, 1, h3, (long long)NP * CSA, CSA);

    // ---- q/k projection (512 -> 128), shared weights, no bias ----
    k_gemm_cl<<<BB * 8 * 8, 128, 0, stream>>>(
        wqkh, 0, CSA, h3, (long long)NP * CSA, CSA, CSA, 8,
        nullptr, 0, 0, xqk, (long long)NP * 128, 128);

    // ---- v projection (512 -> 512), channel-major ----
    k_gemm_xv<<<BB * 32 * 8, 128, 0, stream>>>(wvh, bv, h3, xvt);

    // ---- energy = Xqk . Xqk^T (symmetric, so transposed store == row-major) ----
    k_gemm_cl<<<BB * 128 * 8, 128, 0, stream>>>(
        xqk, (long long)NP * 128, 128, xqk, (long long)NP * 128, 128, 128, 128,
        nullptr, 0, 0, attn, (long long)NP * NP, NP);

    // ---- softmax over keys, then L1 renorm over queries ----
    k_softmax<<<BB * NP, 256, 0, stream>>>(attn);
    k_renorm<<<(BB * NP + 255) / 256, 256, 0, stream>>>(attn);

    // ---- x_r = xv . attn^T fused with hmr = h3 - x_r ----
    k_xr<<<BB * 32 * 8, 128, 0, stream>>>(attn, xvt, h3, hmr);

    // ---- d = wt . hmr + bt (fp32 channel-major) ----
    k_convt<<<BB * 32 * 8, 128, 0, stream>>>(wth, bt, hmr, dbuf);

    // ---- BatchNorm (batch stats) + residual relu add ----
    k_bnstats<<<CSA, 256, 0, stream>>>(dbuf, mu, rsig);
    k_hfin<<<BB * 8 * 32, 256, 0, stream>>>(dbuf, h3, mu, rsig, gam, bet, hfin);

    // ---- conv4 (512 -> 1024) fused with max over N ----
    k_conv4max<<<BB * 64, 128, 0, stream>>>(w4h, b4, hfin, out);

    (void)in_sizes; (void)n_in; (void)out_size; (void)ws_size;
}
